// CocktailGNN_41858751266832
// MI455X (gfx1250) — compile-verified, matrix operand order checked by reading, not run on hardware
//
#include <hip/hip_runtime.h>
#include <hip/hip_bf16.h>

// ---------------------------------------------------------------------------
// GraphSAGE (3x SAGEConv + input MLP) for MI455X / gfx1250, wave32.
//
// Roofline: ~20 GFLOP vs ~4-5 GB of edge gather/scatter traffic -> HBM bound
// (23.3 TB/s). The scatter-mean runs on native global_atomic_add_f32 in fp32.
// The per-layer GEMMs (N x 2F x H) run on the WMMA pipes as bf16-in /
// f32-accumulate (v_wmma_f32_16x16x32_bf16). Weights are pre-packed once per
// call into the exact WMMA B-fragment layout so B loads are global_load_b128;
// activations are loaded as float4 and converted with v_cvt_pk_bf16_f32.
// Each wave owns a 16(M) x 64(N) output strip: one A fragment feeds 4 WMMAs.
// ---------------------------------------------------------------------------

#define GNN_N   50000
#define GNN_E   800000
#define GNN_EMB 128
#define GNN_HID 256

typedef __attribute__((ext_vector_type(16))) __bf16 v16bf;
typedef __attribute__((ext_vector_type(8)))  float  v8f;

// Non-returning fp32 atomic add straight to the L2 atomic units (no CAS loop).
__device__ __forceinline__ void atomic_add_f32(float* p, float v) {
    asm volatile("global_atomic_add_f32 %0, %1, off" : : "v"(p), "v"(v) : "memory");
}

// --------------------------- small utility kernels -------------------------

__global__ __launch_bounds__(256) void zero_f32_kernel(float4* __restrict__ p, int n4) {
    int i = blockIdx.x * blockDim.x + threadIdx.x;
    if (i < n4) p[i] = float4{0.0f, 0.0f, 0.0f, 0.0f};
}

// deg[dst] += 1 for every edge (edge_index row 1 = dst)
__global__ __launch_bounds__(256) void degree_kernel(const int* __restrict__ edge,
                                                     float* __restrict__ deg, int E) {
    int e = blockIdx.x * blockDim.x + threadIdx.x;
    if (e >= E) return;
    atomic_add_f32(deg + edge[E + e], 1.0f);
}

__global__ __launch_bounds__(256) void inv_deg_kernel(const float* __restrict__ deg,
                                                      float* __restrict__ invd, int n) {
    int i = blockIdx.x * blockDim.x + threadIdx.x;
    if (i >= n) return;
    float d = deg[i];
    invd[i] = (d > 0.0f) ? (1.0f / d) : 0.0f;
}

// h0 = relu(x @ W_in + b_in); x:[N,2], W_in:[2,128] row-major. K=2 -> pure VALU.
__global__ __launch_bounds__(256) void input_transform_kernel(
    const float* __restrict__ x, const float* __restrict__ W,
    const float* __restrict__ b, float* __restrict__ h, int Nn) {
    int i = blockIdx.x * blockDim.x + threadIdx.x;   // i = n*128 + c
    if (i >= Nn * GNN_EMB) return;
    int n = i >> 7;
    int c = i & 127;
    float v = fmaf(x[2 * n], W[c], fmaf(x[2 * n + 1], W[GNN_EMB + c], b[c]));
    h[i] = v > 0.0f ? v : 0.0f;
}

// agg[dst] += h[src] for every edge; one thread per (edge, 4-channel chunk).
// lgC = log2(F/4) so edge/channel split is shift/mask (F is 128 or 256).
__global__ __launch_bounds__(256) void scatter_kernel(
    const float* __restrict__ h, const int* __restrict__ edge,
    float* __restrict__ agg, int E, int F, int lgC) {
    int idx = blockIdx.x * blockDim.x + threadIdx.x;
    int total = E << lgC;                     // E * (F/4) <= 51.2M, fits int
    if (idx >= total) return;
    int e = idx >> lgC;
    int c = (idx & ((1 << lgC) - 1)) << 2;
    int s = edge[e];
    int d = edge[E + e];
    const float4 v = *(const float4*)(h + (long long)s * F + c);   // global_load_b128
    float* p = agg + (long long)d * F + c;
    atomic_add_f32(p + 0, v.x);
    atomic_add_f32(p + 1, v.y);
    atomic_add_f32(p + 2, v.z);
    atomic_add_f32(p + 3, v.w);
}

// ------------------------ weight fragment pre-pack -------------------------
// WMMA bf16 B-fragment lane layout (mirror of ISA 7.12.2 16-bit A 16x32):
//   lanes 0-15 : col N=lane,     K in {k0+0..7}  U {k0+16..23}
//   lanes 16-31: col N=lane-16,  K in {k0+8..15} U {k0+24..31}
//   element e (0..15) -> k = kbase + e + (e>=8 ? 8 : 0)
// Packed layout: Wpk[((nt*kblocks + kb)*32 + lane)*16 + e]  (32B per lane rec)
// lgKB = log2(Fin/32) (Fin is 128 or 256 -> kblocks 4 or 8).
__global__ __launch_bounds__(256) void pack_weight_kernel(
    const float* __restrict__ W, __bf16* __restrict__ Wpk,
    int Fin, int Fout, int lgKB) {
    int idx = blockIdx.x * blockDim.x + threadIdx.x;   // (nt*kblocks + kb)*32 + lane
    const int kblocks = Fin >> 5;
    const int total = (Fout >> 4) * kblocks * 32;
    if (idx >= total) return;
    const int lane = idx & 31;
    const int t  = idx >> 5;
    const int kb = t & (kblocks - 1);
    const int nt = t >> lgKB;
    const int n  = (nt << 4) + (lane & 15);
    const int k0 = (kb << 5) + ((lane >> 4) << 3);
    __bf16* dst = Wpk + (size_t)idx * 16;
    #pragma unroll
    for (int e = 0; e < 16; ++e) {
        const int k = k0 + e + ((e >= 8) ? 8 : 0);
        dst[e] = (__bf16)W[k * Fout + n];
    }
}

// ------------------------- fused SAGE layer GEMM ---------------------------
// out = relu( (agg * inv_deg) @ Wl + bl + h @ Wr )
// One wave32 computes a 16(M) x 64(N) strip: 4 f32 accumulator fragments.
// Two K passes (agg/Wl then h/Wr) accumulate into the same fragments.
__global__ __launch_bounds__(256) void sage_gemm_wmma_kernel(
    const float*  __restrict__ h,      // [Nn, Fin]
    const float*  __restrict__ agg,    // [Nn, Fin]
    const float*  __restrict__ invd,   // [Nn]
    const __bf16* __restrict__ Wl_pk,  // packed [Fout/16][Fin/32][32][16]
    const __bf16* __restrict__ Wr_pk,
    const float*  __restrict__ bl,     // [Fout]
    float* __restrict__ out,           // [Nn, Fout]
    int Nn, int Fin, int Fout) {
    const int lane = threadIdx.x & 31;
    const int wave = threadIdx.x >> 5;
    const int ngrp   = Fout >> 6;               // 64-col groups
    const int mtiles = Nn >> 4;
    const int wid = blockIdx.x * 8 + wave;      // wave-uniform
    if (wid >= mtiles * ngrp) return;           // whole-wave exit, EXEC stays all-1s

    const int mt = wid / ngrp;
    const int g  = wid - mt * ngrp;
    const int m0  = mt << 4;
    const int nt0 = g << 2;                     // first of 4 n-tiles
    const int kblocks = Fin >> 5;

    const int r  = lane & 15;
    const int hi = lane >> 4;                   // K-half select
    const int m  = m0 + r;
    const float id = invd[m];

    const size_t frag_stride = (size_t)32 * 16;         // bf16 elems per k-block
    const size_t tile_stride = (size_t)kblocks * frag_stride;
    const size_t lane_off    = (size_t)lane * 16;

    v8f acc0 = {}, acc1 = {}, acc2 = {}, acc3 = {};

    #pragma unroll
    for (int pass = 0; pass < 2; ++pass) {
        const float* __restrict__ Arow =
            (pass ? h : agg) + (size_t)m * Fin + (hi << 3);
        const __bf16* __restrict__ Bp = pass ? Wr_pk : Wl_pk;
        const float scale = pass ? 1.0f : id;

        const __bf16* b0p = Bp + (size_t)(nt0 + 0) * tile_stride + lane_off;
        const __bf16* b1p = Bp + (size_t)(nt0 + 1) * tile_stride + lane_off;
        const __bf16* b2p = Bp + (size_t)(nt0 + 2) * tile_stride + lane_off;
        const __bf16* b3p = Bp + (size_t)(nt0 + 3) * tile_stride + lane_off;

        for (int kb = 0; kb < kblocks; ++kb) {
            // A fragment: two contiguous 8-float runs -> 4x global_load_b128
            const float4 A0 = *(const float4*)(Arow + 0);
            const float4 A1 = *(const float4*)(Arow + 4);
            const float4 A2 = *(const float4*)(Arow + 16);
            const float4 A3 = *(const float4*)(Arow + 20);
            v16bf a;
            a[0]  = (__bf16)(A0.x * scale); a[1]  = (__bf16)(A0.y * scale);
            a[2]  = (__bf16)(A0.z * scale); a[3]  = (__bf16)(A0.w * scale);
            a[4]  = (__bf16)(A1.x * scale); a[5]  = (__bf16)(A1.y * scale);
            a[6]  = (__bf16)(A1.z * scale); a[7]  = (__bf16)(A1.w * scale);
            a[8]  = (__bf16)(A2.x * scale); a[9]  = (__bf16)(A2.y * scale);
            a[10] = (__bf16)(A2.z * scale); a[11] = (__bf16)(A2.w * scale);
            a[12] = (__bf16)(A3.x * scale); a[13] = (__bf16)(A3.y * scale);
            a[14] = (__bf16)(A3.z * scale); a[15] = (__bf16)(A3.w * scale);

            // B fragments: 32B contiguous per lane -> 2x global_load_b128 each
            const v16bf b0 = *(const v16bf*)b0p;
            const v16bf b1 = *(const v16bf*)b1p;
            const v16bf b2 = *(const v16bf*)b2p;
            const v16bf b3 = *(const v16bf*)b3p;

            acc0 = __builtin_amdgcn_wmma_f32_16x16x32_bf16(
                false, a, false, b0, (short)0, acc0, false, false);
            acc1 = __builtin_amdgcn_wmma_f32_16x16x32_bf16(
                false, a, false, b1, (short)0, acc1, false, false);
            acc2 = __builtin_amdgcn_wmma_f32_16x16x32_bf16(
                false, a, false, b2, (short)0, acc2, false, false);
            acc3 = __builtin_amdgcn_wmma_f32_16x16x32_bf16(
                false, a, false, b3, (short)0, acc3, false, false);

            Arow += 32;
            b0p += frag_stride; b1p += frag_stride;
            b2p += frag_stride; b3p += frag_stride;
        }
    }

    // fused bias + ReLU store (C/D layout: lane r = col, VGPR rr = row)
    const int c0 = (nt0 << 4) + r;
    const float bias0 = bl[c0];
    const float bias1 = bl[c0 + 16];
    const float bias2 = bl[c0 + 32];
    const float bias3 = bl[c0 + 48];
    #pragma unroll
    for (int rr = 0; rr < 8; ++rr) {
        const int row = m0 + (hi << 3) + rr;
        float* __restrict__ orow = out + (size_t)row * Fout + c0;
        float v0 = acc0[rr] + bias0;
        float v1 = acc1[rr] + bias1;
        float v2 = acc2[rr] + bias2;
        float v3 = acc3[rr] + bias3;
        orow[0]  = v0 > 0.0f ? v0 : 0.0f;
        orow[16] = v1 > 0.0f ? v1 : 0.0f;
        orow[32] = v2 > 0.0f ? v2 : 0.0f;
        orow[48] = v3 > 0.0f ? v3 : 0.0f;
    }
}

// --------------------------------- launcher --------------------------------

static inline int ceil_div(int a, int b) { return (a + b - 1) / b; }
static inline int ilog2(int v) { int s = 0; while ((1 << s) < v) ++s; return s; }

extern "C" void kernel_launch(void* const* d_in, const int* in_sizes, int n_in,
                              void* d_out, int out_size, void* d_ws, size_t ws_size,
                              hipStream_t stream) {
    (void)in_sizes; (void)n_in; (void)out_size; (void)ws_size;

    const float* x    = (const float*)d_in[0];
    const int*   edge = (const int*)  d_in[1];   // [2, E]: row0=src, row1=dst
    const float* W_in = (const float*)d_in[2];
    const float* b_in = (const float*)d_in[3];
    const float* Wl1  = (const float*)d_in[4];
    const float* bl1  = (const float*)d_in[5];
    const float* Wr1  = (const float*)d_in[6];
    const float* Wl2  = (const float*)d_in[7];
    const float* bl2  = (const float*)d_in[8];
    const float* Wr2  = (const float*)d_in[9];
    const float* Wl3  = (const float*)d_in[10];
    const float* bl3  = (const float*)d_in[11];
    const float* Wr3  = (const float*)d_in[12];
    float* out = (float*)d_out;

    const int N = GNN_N, E = GNN_E;

    // workspace layout (fp32): bufA[N*256] | bufB[N*256] | agg[N*256] | deg[N]
    //                          | invd[N] | 6x packed bf16 weights
    float* bufA = (float*)d_ws;
    float* bufB = bufA + (size_t)N * GNN_HID;
    float* agg  = bufB + (size_t)N * GNN_HID;
    float* deg  = agg  + (size_t)N * GNN_HID;
    float* invd = deg  + N;
    __bf16* pk  = (__bf16*)(invd + N);
    const size_t szL1 = (size_t)GNN_EMB * GNN_HID;   // 128*256
    const size_t szL2 = (size_t)GNN_HID * GNN_HID;   // 256*256
    __bf16* Wl1p = pk;             __bf16* Wr1p = Wl1p + szL1;
    __bf16* Wl2p = Wr1p + szL1;    __bf16* Wr2p = Wl2p + szL2;
    __bf16* Wl3p = Wr2p + szL2;    __bf16* Wr3p = Wl3p + szL2;

    const int TB = 256;

    // --- pack all six weight matrices into WMMA fragment order ---
    {
        const int t1 = (GNN_HID / 16) * (GNN_EMB / 32) * 32;   // 2048
        const int t2 = (GNN_HID / 16) * (GNN_HID / 32) * 32;   // 4096
        const int lg1 = ilog2(GNN_EMB / 32), lg2 = ilog2(GNN_HID / 32);
        pack_weight_kernel<<<ceil_div(t1, TB), TB, 0, stream>>>(Wl1, Wl1p, GNN_EMB, GNN_HID, lg1);
        pack_weight_kernel<<<ceil_div(t1, TB), TB, 0, stream>>>(Wr1, Wr1p, GNN_EMB, GNN_HID, lg1);
        pack_weight_kernel<<<ceil_div(t2, TB), TB, 0, stream>>>(Wl2, Wl2p, GNN_HID, GNN_HID, lg2);
        pack_weight_kernel<<<ceil_div(t2, TB), TB, 0, stream>>>(Wr2, Wr2p, GNN_HID, GNN_HID, lg2);
        pack_weight_kernel<<<ceil_div(t2, TB), TB, 0, stream>>>(Wl3, Wl3p, GNN_HID, GNN_HID, lg2);
        pack_weight_kernel<<<ceil_div(t2, TB), TB, 0, stream>>>(Wr3, Wr3p, GNN_HID, GNN_HID, lg2);
    }

    // --- degrees / mean normalizer (once; reused by all 3 layers) ---
    zero_f32_kernel<<<ceil_div(N / 4, TB), TB, 0, stream>>>((float4*)deg, N / 4);
    degree_kernel<<<ceil_div(E, TB), TB, 0, stream>>>(edge, deg, E);
    inv_deg_kernel<<<ceil_div(N, TB), TB, 0, stream>>>(deg, invd, N);

    // --- h0 = relu(x @ W_in + b_in) -> bufA [N,128] ---
    input_transform_kernel<<<ceil_div(N * GNN_EMB, TB), TB, 0, stream>>>(
        x, W_in, b_in, bufA, N);

    // --- layer 1: 128 -> 256, bufA -> bufB ---
    {
        const int Fin = GNN_EMB, Fout = GNN_HID;
        const int lgC = ilog2(Fin / 4);
        zero_f32_kernel<<<ceil_div(N * Fin / 4, TB), TB, 0, stream>>>(
            (float4*)agg, N * Fin / 4);
        scatter_kernel<<<ceil_div(E * (Fin / 4), TB), TB, 0, stream>>>(
            bufA, edge, agg, E, Fin, lgC);
        const int waves = (N / 16) * (Fout / 64);
        sage_gemm_wmma_kernel<<<ceil_div(waves, 8), TB, 0, stream>>>(
            bufA, agg, invd, Wl1p, Wr1p, bl1, bufB, N, Fin, Fout);
    }

    // --- layer 2: 256 -> 256, bufB -> bufA ---
    {
        const int Fin = GNN_HID, Fout = GNN_HID;
        const int lgC = ilog2(Fin / 4);
        zero_f32_kernel<<<ceil_div(N * Fin / 4, TB), TB, 0, stream>>>(
            (float4*)agg, N * Fin / 4);
        scatter_kernel<<<ceil_div(E * (Fin / 4), TB), TB, 0, stream>>>(
            bufB, edge, agg, E, Fin, lgC);
        const int waves = (N / 16) * (Fout / 64);
        sage_gemm_wmma_kernel<<<ceil_div(waves, 8), TB, 0, stream>>>(
            bufB, agg, invd, Wl2p, Wr2p, bl2, bufA, N, Fin, Fout);
    }

    // --- layer 3: 256 -> 256, bufA -> d_out ---
    {
        const int Fin = GNN_HID, Fout = GNN_HID;
        const int lgC = ilog2(Fin / 4);
        zero_f32_kernel<<<ceil_div(N * Fin / 4, TB), TB, 0, stream>>>(
            (float4*)agg, N * Fin / 4);
        scatter_kernel<<<ceil_div(E * (Fin / 4), TB), TB, 0, stream>>>(
            bufA, edge, agg, E, Fin, lgC);
        const int waves = (N / 16) * (Fout / 64);
        sage_gemm_wmma_kernel<<<ceil_div(waves, 8), TB, 0, stream>>>(
            bufA, agg, invd, Wl3p, Wr3p, bl3, out, N, Fin, Fout);
    }
}
